// MambaWrapper_69363721830796
// MI455X (gfx1250) — matrix operand order repaired and checked by reading, not compiled, not run on hardware
//
#include <hip/hip_runtime.h>
#include <hip/hip_bf16.h>
#include <math.h>

#define BATCH   2
#define SEQ     2048
#define D_MODEL 1024
#define D_STATE 16
#define D_CONV  4
#define D_INNER 2048
#define DT_RANK 64
#define M_ROWS  (BATCH * SEQ)   // 4096

typedef __attribute__((ext_vector_type(16))) __bf16 v16bf;
typedef __attribute__((ext_vector_type(8)))  __bf16 v8bf;
typedef __attribute__((ext_vector_type(8)))  float  v8f;

// ---------------------------------------------------------------------------
// helpers
// ---------------------------------------------------------------------------
__device__ __forceinline__ float softplusf(float x) {
  return (x > 20.f) ? x : log1pf(__expf(x));
}
__device__ __forceinline__ float sigmoidf_(float x) {
  return 1.f / (1.f + __expf(-x));
}

// A fragment (16x32 bf16): lane (m, g) holds K = 16g .. 16g+15
__device__ __forceinline__ v16bf ld_frag_a(const __bf16* __restrict__ row,
                                           int k0, int g) {
  v8bf lo = *reinterpret_cast<const v8bf*>(row + k0 + 16 * g);
  v8bf hi = *reinterpret_cast<const v8bf*>(row + k0 + 16 * g + 8);
  v16bf r;
  #pragma unroll
  for (int i = 0; i < 8; ++i) { r[i] = lo[i]; r[8 + i] = hi[i]; }
  return r;
}
// B fragment (32x16 = W^T): lane (n, g) holds K = 16g .. 16g+15 (rowpg has +16g)
__device__ __forceinline__ v16bf ld_frag_b(const __bf16* __restrict__ rowpg,
                                           int k0) {
  v8bf lo = *reinterpret_cast<const v8bf*>(rowpg + k0);
  v8bf hi = *reinterpret_cast<const v8bf*>(rowpg + k0 + 8);
  v16bf r;
  #pragma unroll
  for (int i = 0; i < 8; ++i) { r[i] = lo[i]; r[8 + i] = hi[i]; }
  return r;
}

// ---------------------------------------------------------------------------
// Big WMMA GEMM: C[M,N](f32) = A[M,K](bf16) @ W[N,K](bf16)^T
// Block tile 64(M) x 256(N); 8 waves as 2(M) x 4(N); wave tile 32 x 64
// (2 A-frags x 4 B-frags -> 8 v_wmma per 32-wide k-step).
// EPI 0: plain store.  EPI 1: softplus(c + bias[n]).
// Grid: (M/64, N/256); M, N, K must be multiples of 64/256/32.
// ---------------------------------------------------------------------------
template<int EPI>
__global__ __launch_bounds__(256) void gemm_bf16_wmma_big(
    const __bf16* __restrict__ A, int lda,
    const __bf16* __restrict__ W, int ldb,
    float* __restrict__ C, int ldc, int K,
    const float* __restrict__ bias)
{
  const int lane = threadIdx.x & 31;
  const int wave = threadIdx.x >> 5;
  const int wm   = wave >> 2;                       // 0..1
  const int wn   = wave & 3;                        // 0..3
  const int m0   = blockIdx.x * 64 + wm * 32;
  const int n0   = blockIdx.y * 256 + wn * 64;
  const int lh   = lane & 15;
  const int g    = lane >> 4;

  const __bf16* __restrict__ ar[2];
  ar[0] = A + (size_t)(m0 + lh) * lda;
  ar[1] = A + (size_t)(m0 + 16 + lh) * lda;
  const __bf16* __restrict__ br[4];
  #pragma unroll
  for (int j = 0; j < 4; ++j)
    br[j] = W + (size_t)(n0 + j * 16 + lh) * ldb + 16 * g;

  v8f acc[2][4];
  #pragma unroll
  for (int i = 0; i < 2; ++i)
    #pragma unroll
    for (int j = 0; j < 4; ++j) acc[i][j] = {};

  for (int k0 = 0; k0 < K; k0 += 32) {
    // prefetch next k-slab (global_prefetch_b8)
    __builtin_prefetch(ar[0] + k0 + 64, 0, 0);
    __builtin_prefetch(br[0] + k0 + 64, 0, 0);

    v16bf af[2], bf[4];
    #pragma unroll
    for (int i = 0; i < 2; ++i) af[i] = ld_frag_a(ar[i], k0, g);
    #pragma unroll
    for (int j = 0; j < 4; ++j) bf[j] = ld_frag_b(br[j], k0);

    #pragma unroll
    for (int i = 0; i < 2; ++i)
      #pragma unroll
      for (int j = 0; j < 4; ++j)
        acc[i][j] = __builtin_amdgcn_wmma_f32_16x16x32_bf16(
            false, af[i], false, bf[j], (short)0, acc[i][j], false, false);
  }

  #pragma unroll
  for (int i = 0; i < 2; ++i) {
    const int mb = m0 + i * 16 + 8 * g;     // VGPR v -> row mb + v
    #pragma unroll
    for (int j = 0; j < 4; ++j) {
      const int n = n0 + j * 16 + lh;
      const float bv = (EPI == 1) ? bias[n] : 0.f;
      #pragma unroll
      for (int v = 0; v < 8; ++v) {
        float val = acc[i][j][v];
        if (EPI == 1) val = softplusf(val + bv);
        C[(size_t)(mb + v) * ldc + n] = val;
      }
    }
  }
}

// ---------------------------------------------------------------------------
// Small WMMA GEMM (one 16x16 tile per wave) for N not divisible by 256 (x_proj)
// ---------------------------------------------------------------------------
__global__ __launch_bounds__(256) void gemm_bf16_wmma_small(
    const __bf16* __restrict__ A, int lda,
    const __bf16* __restrict__ W, int ldb,
    float* __restrict__ C, int ldc,
    int M, int N, int K)
{
  const int lane   = threadIdx.x & 31;
  const int wave   = threadIdx.x >> 5;
  const int ntiles = N >> 4;
  const int tile   = blockIdx.x * 8 + wave;
  if (tile >= (M >> 4) * ntiles) return;   // wave-uniform; EXEC stays full
  const int mt = tile / ntiles;
  const int nt = tile - mt * ntiles;
  const int lh = lane & 15;
  const int g  = lane >> 4;

  const __bf16* __restrict__ arow = A + (size_t)(mt * 16 + lh) * lda;
  const __bf16* __restrict__ brow = W + (size_t)(nt * 16 + lh) * ldb + 16 * g;

  v8f acc = {};
  for (int k0 = 0; k0 < K; k0 += 32) {
    v16bf af = ld_frag_a(arow, k0, g);
    v16bf bf = ld_frag_b(brow, k0);
    acc = __builtin_amdgcn_wmma_f32_16x16x32_bf16(
        false, af, false, bf, (short)0, acc, false, false);
  }
  const int n  = nt * 16 + lh;
  const int mb = mt * 16 + 8 * g;
  #pragma unroll
  for (int v = 0; v < 8; ++v)
    C[(size_t)(mb + v) * ldc + n] = acc[v];
}

// ---------------------------------------------------------------------------
// fp32 -> bf16 bulk convert (vectorized x4); n must be a multiple of 4
// ---------------------------------------------------------------------------
__global__ __launch_bounds__(256) void cvt_f32_bf16_kernel(
    const float* __restrict__ in, __bf16* __restrict__ out, int n4)
{
  const int i = blockIdx.x * blockDim.x + threadIdx.x;
  if (i >= n4) return;
  float4 v = reinterpret_cast<const float4*>(in)[i];
  __bf16 o[4] = {(__bf16)v.x, (__bf16)v.y, (__bf16)v.z, (__bf16)v.w};
  reinterpret_cast<uint64_t*>(out)[i] = *reinterpret_cast<uint64_t*>(o);
}

// Extract dt columns (0..63) of xdbl[4096,96] into dense bf16 [4096,64]
__global__ __launch_bounds__(256) void extract_dt_bf16_kernel(
    const float* __restrict__ xdbl, __bf16* __restrict__ dtb)
{
  const int idx = blockIdx.x * blockDim.x + threadIdx.x;  // M_ROWS*DT_RANK
  if (idx >= M_ROWS * DT_RANK) return;
  const int m = idx >> 6;
  const int c = idx & 63;
  dtb[idx] = (__bf16)xdbl[(size_t)m * 96 + c];
}

// ---------------------------------------------------------------------------
// Depthwise causal conv1d + bias + SiLU; reads xz fp32, writes bf16 u.
// ---------------------------------------------------------------------------
__global__ __launch_bounds__(256) void conv_silu_kernel(
    const float* __restrict__ xz,      // [M_ROWS, 2*D_INNER]
    const float* __restrict__ conv_w,  // [D_INNER, D_CONV]
    const float* __restrict__ conv_b,  // [D_INNER]
    __bf16* __restrict__ u)            // [M_ROWS, D_INNER] bf16
{
  const int idx = blockIdx.x * blockDim.x + threadIdx.x;
  if (idx >= M_ROWS * D_INNER) return;
  const int d = idx & (D_INNER - 1);
  const int m = idx >> 11;
  const int t = m & (SEQ - 1);
  float acc = conv_b[d];
  #pragma unroll
  for (int j = 0; j < D_CONV; ++j) {
    const int tt = t - (D_CONV - 1) + j;
    if (tt >= 0)
      acc += conv_w[d * D_CONV + j] *
             xz[(size_t)(m - (D_CONV - 1) + j) * (2 * D_INNER) + d];
  }
  u[(size_t)m * D_INNER + d] = (__bf16)(acc * sigmoidf_(acc));
}

// ---------------------------------------------------------------------------
// Selective scan fused with D-skip + SiLU(z) gate. One lane per (b, channel),
// h[16] state in registers.  B_t||C_t (32 contiguous floats of the xdbl row)
// is double-buffered in LDS via global_load_async_to_lds_b128: while step t
// computes, step t+1's operands stream into the alternate buffer (ASYNCcnt),
// taking the global-load latency off the recurrence critical path.
// ---------------------------------------------------------------------------
__global__ __launch_bounds__(256) void scan_fused_kernel(
    const float* __restrict__ xz,      // z at column D_INNER + d
    const __bf16* __restrict__ u,      // [M_ROWS, D_INNER] bf16
    const float* __restrict__ xdbl,    // [M_ROWS, 96]  (dt | B | C)
    const float* __restrict__ delta,   // [M_ROWS, D_INNER]
    const float* __restrict__ A_log,   // [D_INNER, D_STATE]
    const float* __restrict__ Dp,      // [D_INNER]
    __bf16* __restrict__ yg)           // [M_ROWS, D_INNER] bf16
{
  const int b = blockIdx.y;
  const int d = blockIdx.x * 256 + threadIdx.x;

  float Av[D_STATE];
  #pragma unroll
  for (int n = 0; n < D_STATE; ++n) Av[n] = -__expf(A_log[d * D_STATE + n]);
  const float Dd = Dp[d];
  float h[D_STATE];
  #pragma unroll
  for (int n = 0; n < D_STATE; ++n) h[n] = 0.f;

  // double-buffered B||C: sBC[phase][0..15]=B_t, [16..31]=C_t
  __shared__ float sBC[2][2 * D_STATE];

  const int tid = threadIdx.x;
  // prologue: async-load t=0 into buffer 0 (8 lanes x b128 = 128 B)
  if (tid < 8) {
    unsigned lds_a = (unsigned)(uintptr_t)(&sBC[0][tid * 4]);
    const float* ga = xdbl + (size_t)b * SEQ * 96 + DT_RANK + tid * 4;
    asm volatile("global_load_async_to_lds_b128 %0, %1, off"
                 :: "v"(lds_a), "v"(ga) : "memory");
  }
  asm volatile("s_wait_asynccnt 0x0" ::: "memory");
  __syncthreads();

  int ph = 0;
  for (int t = 0; t < SEQ; ++t) {
    const size_t m = (size_t)b * SEQ + t;

    // kick off async prefetch of step t+1 into the other buffer
    if (t + 1 < SEQ && tid < 8) {
      unsigned lds_a = (unsigned)(uintptr_t)(&sBC[ph ^ 1][tid * 4]);
      const float* ga = xdbl + (m + 1) * 96 + DT_RANK + tid * 4;
      asm volatile("global_load_async_to_lds_b128 %0, %1, off"
                   :: "v"(lds_a), "v"(ga) : "memory");
    }

    const float dt = delta[m * D_INNER + d];
    const float ut = (float)u[m * D_INNER + d];
    const float du = dt * ut;
    float y = 0.f;
    #pragma unroll
    for (int n = 0; n < D_STATE; ++n) {
      const float dA = __expf(dt * Av[n]);
      h[n] = dA * h[n] + du * sBC[ph][n];
      y += h[n] * sBC[ph][D_STATE + n];
    }
    const float zt = xz[m * (2 * D_INNER) + D_INNER + d];
    yg[m * D_INNER + d] = (__bf16)((y + ut * Dd) * (zt * sigmoidf_(zt)));

    // next-step operands must have landed; flip buffers
    asm volatile("s_wait_asynccnt 0x0" ::: "memory");
    __syncthreads();
    ph ^= 1;
  }
}

// ---------------------------------------------------------------------------
extern "C" void kernel_launch(void* const* d_in, const int* in_sizes, int n_in,
                              void* d_out, int out_size, void* d_ws, size_t ws_size,
                              hipStream_t stream) {
  const float* x          = (const float*)d_in[0];
  const float* in_proj_w  = (const float*)d_in[1];
  const float* conv_w     = (const float*)d_in[2];
  const float* conv_b     = (const float*)d_in[3];
  const float* x_proj_w   = (const float*)d_in[4];
  const float* dt_proj_w  = (const float*)d_in[5];
  const float* dt_proj_b  = (const float*)d_in[6];
  const float* A_log      = (const float*)d_in[7];
  const float* Dp         = (const float*)d_in[8];
  const float* out_proj_w = (const float*)d_in[9];
  float* out = (float*)d_out;

  // ---- workspace layout (bytes, 16B-aligned chunks) ----
  char* p = (char*)d_ws;
  float*  xz     = (float*)p;  p += (size_t)M_ROWS * 2 * D_INNER * 4;  // 64 MB
  __bf16* u_bf   = (__bf16*)p; p += (size_t)M_ROWS * D_INNER * 2;      // 16 MB
  float*  xdbl   = (float*)p;  p += (size_t)M_ROWS * 96 * 4;           // 1.5 MB
  __bf16* dt_bf  = (__bf16*)p; p += (size_t)M_ROWS * DT_RANK * 2;      // 0.5 MB
  float*  delta  = (float*)p;  p += (size_t)M_ROWS * D_INNER * 4;      // 32 MB
  __bf16* yg_bf  = (__bf16*)p; p += (size_t)M_ROWS * D_INNER * 2;      // 16 MB
  __bf16* x_bf   = (__bf16*)p; p += (size_t)M_ROWS * D_MODEL * 2;      // 8 MB
  __bf16* win_bf = (__bf16*)p; p += (size_t)2 * D_INNER * D_MODEL * 2; // 8 MB
  __bf16* wxp_bf = (__bf16*)p; p += (size_t)96 * D_INNER * 2;          // .375 MB
  __bf16* wdt_bf = (__bf16*)p; p += (size_t)D_INNER * DT_RANK * 2;     // .25 MB
  __bf16* wop_bf = (__bf16*)p; p += (size_t)D_MODEL * D_INNER * 2;     // 4 MB

  auto cvt = [&](const float* src, __bf16* dst, size_t n) {
    const int n4 = (int)(n / 4);
    cvt_f32_bf16_kernel<<<(n4 + 255) / 256, 256, 0, stream>>>(src, dst, n4);
  };

  // 0) one-shot bf16 conversions (activations/weights)
  cvt(x,          x_bf,   (size_t)M_ROWS * D_MODEL);
  cvt(in_proj_w,  win_bf, (size_t)2 * D_INNER * D_MODEL);
  cvt(x_proj_w,   wxp_bf, (size_t)96 * D_INNER);
  cvt(dt_proj_w,  wdt_bf, (size_t)D_INNER * DT_RANK);
  cvt(out_proj_w, wop_bf, (size_t)D_MODEL * D_INNER);

  // 1) in_proj: xz = x @ in_proj_w^T   (M=4096, N=4096, K=1024)
  {
    dim3 grid(M_ROWS / 64, (2 * D_INNER) / 256);
    gemm_bf16_wmma_big<0><<<grid, 256, 0, stream>>>(
        x_bf, D_MODEL, win_bf, D_MODEL, xz, 2 * D_INNER, D_MODEL, nullptr);
  }
  // 2) depthwise causal conv + SiLU -> u (bf16)
  conv_silu_kernel<<<(M_ROWS * D_INNER) / 256, 256, 0, stream>>>(
      xz, conv_w, conv_b, u_bf);
  // 3) x_proj: xdbl = u @ x_proj_w^T   (M=4096, N=96, K=2048)
  {
    const int tiles = (M_ROWS / 16) * (96 / 16);
    gemm_bf16_wmma_small<<<(tiles + 7) / 8, 256, 0, stream>>>(
        u_bf, D_INNER, wxp_bf, D_INNER, xdbl, 96, M_ROWS, 96, D_INNER);
  }
  // 3b) repack dt columns to dense bf16 for the big GEMM (lda=64)
  extract_dt_bf16_kernel<<<(M_ROWS * DT_RANK) / 256, 256, 0, stream>>>(
      xdbl, dt_bf);
  // 4) dt_proj + softplus: delta = softplus(dt @ dt_proj_w^T + b)
  {
    dim3 grid(M_ROWS / 64, D_INNER / 256);
    gemm_bf16_wmma_big<1><<<grid, 256, 0, stream>>>(
        dt_bf, DT_RANK, wdt_bf, DT_RANK, delta, D_INNER, DT_RANK, dt_proj_b);
  }
  // 5) selective scan + skip + gate -> yg (bf16), async-prefetched B/C
  {
    dim3 grid(D_INNER / 256, BATCH);
    scan_fused_kernel<<<grid, 256, 0, stream>>>(
        xz, u_bf, xdbl, delta, A_log, Dp, yg_bf);
  }
  // 6) out_proj: out = yg @ out_proj_w^T  (M=4096, N=1024, K=2048)
  {
    dim3 grid(M_ROWS / 64, D_MODEL / 256);
    gemm_bf16_wmma_big<0><<<grid, 256, 0, stream>>>(
        yg_bf, D_INNER, wop_bf, D_INNER, out, D_MODEL, D_INNER, nullptr);
  }
}
